// SNNet_28278064677060
// MI455X (gfx1250) — compile-verified, matrix operand order checked by reading, not run on hardware
//
#include <hip/hip_runtime.h>

// ---------------------------------------------------------------------------
// SNN (3-layer LIF, 25 steps) for MI455X / gfx1250, wave32 + WMMA bf16 + TDM.
//
//   1) split f32 -> bf16 hi/lo for x, W1, W2 (hi+lo sum ~f32-accurate)
//   2) GEMM1: cur1 = x@W1^T + b1            (3-product bf16 WMMA, LDS via ds path)
//   3) LIF1 scan: spk1[t] for all 25 t      (layer1 depends only on cur1)
//   4) GEMM2 batched over t: cur2[t] = spk1[t]@W2^T + b2
//      - heavy op (215 GFLOP): 2-product bf16 WMMA (spikes exact in bf16)
//      - weight panels staged in LDS by the Tensor Data Mover, double-buffered
//        (tensor_load_to_lds + s_wait_tensorcnt), falling back to ds_store
//        staging when the TDM builtin is unavailable.
//   5) LIF2 scan: spk2[t]
//   6) head: per-batch dot with Wout (O=2) + output LIF scan
// ---------------------------------------------------------------------------

typedef __attribute__((ext_vector_type(16))) __bf16 v16bf;
typedef __attribute__((ext_vector_type(8)))  __bf16 v8bf;
typedef __attribute__((ext_vector_type(8)))  float  v8f;

constexpr int BB  = 1024;   // batch
constexpr int DIN = 2048;
constexpr int H1  = 4096;
constexpr int H2  = 1024;
constexpr int NT  = 25;     // num_steps (from setup_inputs)
constexpr int LDB = 40;     // padded LDS row stride in bf16 elems (80 B, 16B-aligned)

#if __has_builtin(__builtin_amdgcn_tensor_load_to_lds)
#define USE_TDM 1
#else
#define USE_TDM 0
#endif

// ---------------------------------------------------------------- split f32 -> hi/lo bf16
__global__ void split_hilo_kernel(const float* __restrict__ src,
                                  __bf16* __restrict__ hi, __bf16* __restrict__ lo, int n) {
    int i = blockIdx.x * blockDim.x + threadIdx.x;
    if (i < n) {
        float v = src[i];
        __bf16 h = (__bf16)v;
        hi[i] = h;
        lo[i] = (__bf16)(v - (float)h);
    }
}

__device__ inline v16bf frag16(const __bf16* p0, const __bf16* p1) {
    v8bf a = *(const v8bf*)p0;
    v8bf b = *(const v8bf*)p1;
    return __builtin_shufflevector(a, b, 0,1,2,3,4,5,6,7,8,9,10,11,12,13,14,15);
}

#if USE_TDM
typedef __attribute__((ext_vector_type(4))) unsigned int u32x4;
typedef __attribute__((ext_vector_type(8))) int          i32x8;
typedef __attribute__((ext_vector_type(4))) int          i32x4;

// On CDNA5 a generic pointer into LDS carries the LDS byte address in its low
// 32 bits (aperture check discards the upper bits), which is what D#.lds_addr wants.
__device__ inline unsigned int lds_addr_of(const void* p) {
    return (unsigned int)(uintptr_t)p;
}

// Issue one TDM DMA: 2D tile of `rows` x 32 bf16 elements from a row-major
// [*, K] bf16 matrix into LDS with hardware row padding to LDB=40 elements.
//   pad: tile row = 32 bf16 = 64 B = 16 DWORDs -> pad_interval enc 3 (16 DW),
//        pad 8 bf16 = 4 DWORDs -> pad_amount enc 3.
__device__ inline void tdm_load_panel(unsigned int ldsAddr, const __bf16* gsrc,
                                      int K, int rows) {
    unsigned long long ga = (unsigned long long)(uintptr_t)gsrc;
    u32x4 g0;
    g0[0] = 1u;                                               // count=1, user D#
    g0[1] = ldsAddr;                                          // lds_addr
    g0[2] = (unsigned int)ga;                                 // global_addr[31:0]
    g0[3] = (unsigned int)((ga >> 32) & 0x1FFFFFFu) | (2u << 30);  // addr[56:32] | type=2
    i32x8 g1;
    g1[0] = (1 << 16) | (1 << 20) | (3 << 22) | (3 << 25);    // data_size=2B, pad_en, intv=16DW, amt=4DW
    g1[1] = (int)((unsigned)(K & 0xFFFF) << 16);              // tensor_dim0[15:0]
    g1[2] = (int)(((unsigned)K >> 16) | ((unsigned)(rows & 0xFFFF) << 16)); // dim0 hi | dim1 lo
    g1[3] = (int)(((unsigned)rows >> 16) | (32u << 16));      // dim1 hi | tile_dim0=32
    g1[4] = rows;                                             // tile_dim1=rows, tile_dim2=0
    g1[5] = K;                                                // tensor_dim0_stride[31:0]
    g1[6] = 0;                                                // stride hi | dim1_stride lo
    g1[7] = 0;
    i32x4 gz4 = {0, 0, 0, 0};                                 // 2D tensor: groups 2/3 zero
    i32x8 gz8 = {0, 0, 0, 0, 0, 0, 0, 0};
    // 6-arg form (amdgpu-toolchain / clang-23 flavor on this driver):
    //   (u32x4 g0, i32x8 g1, i32x4, i32x4, i32x8, i32 cpol)
    __builtin_amdgcn_tensor_load_to_lds(g0, g1, gz4, gz4, gz8, 0);
}
#endif

// ---------------------------------------------------------------- GEMM1 (3 products)
// C[M,N] = Ahi*Bhi + Ahi*Blo + Alo*Bhi + bias ;  A:[M,K] row-major, B:[N,K] row-major (=W^T)
__global__ __launch_bounds__(256)
void gemm_hilo3_kernel(const __bf16* __restrict__ Ahi, const __bf16* __restrict__ Alo,
                       const __bf16* __restrict__ Bhi, const __bf16* __restrict__ Blo,
                       const float* __restrict__ bias, float* __restrict__ C,
                       int M, int N, int K) {
    __shared__ __align__(16) __bf16 sBh[64 * LDB];
    __shared__ __align__(16) __bf16 sBl[64 * LDB];
    const int nBase  = blockIdx.x * 64;
    const int mBase  = blockIdx.y * 128;
    const int wave   = threadIdx.x >> 5;
    const int lane   = threadIdx.x & 31;
    const int laneN  = lane & 15;
    const int hiHalf = lane >> 4;
    const __bf16* ah = Ahi + (size_t)(mBase + wave * 16 + laneN) * K;
    const __bf16* al = Alo + (size_t)(mBase + wave * 16 + laneN) * K;
    const int ldRow = threadIdx.x >> 2;         // 0..63
    const int ldCol = (threadIdx.x & 3) * 8;    // 0,8,16,24
    const __bf16* gBh = Bhi + (size_t)(nBase + ldRow) * K + ldCol;
    const __bf16* gBl = Blo + (size_t)(nBase + ldRow) * K + ldCol;

    v8f acc[4];
    v8f zero = {0.f,0.f,0.f,0.f,0.f,0.f,0.f,0.f};
#pragma unroll
    for (int i = 0; i < 4; ++i) acc[i] = zero;

    for (int kc = 0; kc < K; kc += 32) {
        *(v8bf*)&sBh[ldRow * LDB + ldCol] = *(const v8bf*)(gBh + kc);
        *(v8bf*)&sBl[ldRow * LDB + ldCol] = *(const v8bf*)(gBl + kc);
        __syncthreads();
        const int kb0 = kc + hiHalf * 8, kb1 = kc + 16 + hiHalf * 8;
        v16bf afh = frag16(ah + kb0, ah + kb1);
        v16bf afl = frag16(al + kb0, al + kb1);
        if (kc + 32 < K) __builtin_prefetch(ah + kc + 32, 0, 0);
        const int lb0 = hiHalf * 8, lb1 = 16 + hiHalf * 8;
#pragma unroll
        for (int nt = 0; nt < 4; ++nt) {
            const __bf16* rh = &sBh[(nt * 16 + laneN) * LDB];
            const __bf16* rl = &sBl[(nt * 16 + laneN) * LDB];
            v16bf bh = frag16(rh + lb0, rh + lb1);
            v16bf bl = frag16(rl + lb0, rl + lb1);
            acc[nt] = __builtin_amdgcn_wmma_f32_16x16x32_bf16(false, afh, false, bh,
                                                              (short)0, acc[nt], false, false);
            acc[nt] = __builtin_amdgcn_wmma_f32_16x16x32_bf16(false, afh, false, bl,
                                                              (short)0, acc[nt], false, false);
            acc[nt] = __builtin_amdgcn_wmma_f32_16x16x32_bf16(false, afl, false, bh,
                                                              (short)0, acc[nt], false, false);
        }
        __syncthreads();
    }
#pragma unroll
    for (int nt = 0; nt < 4; ++nt) {
        const int n = nBase + nt * 16 + laneN;
        const float bv = bias[n];
#pragma unroll
        for (int r = 0; r < 8; ++r) {
            const int m = mBase + wave * 16 + hiHalf * 8 + r;
            C[(size_t)m * N + n] = acc[nt][r] + bv;
        }
    }
}

// ---------------------------------------------------------------- GEMM2 (2 products, batched over t)
// C[t][M,N] = A[t]*Bhi + A[t]*Blo + bias ; A[t] = spikes (exact in bf16)
__global__ __launch_bounds__(256)
void gemm_spk2_kernel(const __bf16* __restrict__ Aall,
                      const __bf16* __restrict__ Bhi, const __bf16* __restrict__ Blo,
                      const float* __restrict__ bias, float* __restrict__ Call,
                      int M, int N, int K) {
    const int t      = blockIdx.z;
    const int nBase  = blockIdx.x * 64;
    const int mBase  = blockIdx.y * 128;
    const int wave   = threadIdx.x >> 5;
    const int lane   = threadIdx.x & 31;
    const int laneN  = lane & 15;
    const int hiHalf = lane >> 4;
    const __bf16* A = Aall + (size_t)t * M * K + (size_t)(mBase + wave * 16 + laneN) * K;

    v8f acc[4];
    v8f zero = {0.f,0.f,0.f,0.f,0.f,0.f,0.f,0.f};
#pragma unroll
    for (int i = 0; i < 4; ++i) acc[i] = zero;

#if USE_TDM
    // Double-buffered LDS panels filled by the Tensor Data Mover.
    __shared__ __align__(16) __bf16 sB[2][2][64 * LDB];
    const __bf16* gBhBase = Bhi + (size_t)nBase * K;
    const __bf16* gBlBase = Blo + (size_t)nBase * K;
    if (wave == 0) {                                  // TDM ignores EXEC; one wave issues
        tdm_load_panel(lds_addr_of(&sB[0][0][0]), gBhBase, K, 64);
        tdm_load_panel(lds_addr_of(&sB[0][1][0]), gBlBase, K, 64);
    }
    for (int kc = 0; kc < K; kc += 32) {
        const int cur = (kc >> 5) & 1;
        const bool hasNext = (kc + 32 < K);
        if (wave == 0) {
            if (hasNext) {
                tdm_load_panel(lds_addr_of(&sB[1 - cur][0][0]), gBhBase + kc + 32, K, 64);
                tdm_load_panel(lds_addr_of(&sB[1 - cur][1][0]), gBlBase + kc + 32, K, 64);
                __builtin_amdgcn_s_wait_tensorcnt(2);   // current panel done; next in flight
            } else {
                __builtin_amdgcn_s_wait_tensorcnt(0);
            }
        }
        __syncthreads();                                // current panel visible to all waves
        const int kb0 = kc + hiHalf * 8, kb1 = kc + 16 + hiHalf * 8;
        v16bf af = frag16(A + kb0, A + kb1);
        if (hasNext) __builtin_prefetch(A + kc + 32, 0, 0);
        const int lb0 = hiHalf * 8, lb1 = 16 + hiHalf * 8;
#pragma unroll
        for (int nt = 0; nt < 4; ++nt) {
            const __bf16* rh = &sB[cur][0][(nt * 16 + laneN) * LDB];
            const __bf16* rl = &sB[cur][1][(nt * 16 + laneN) * LDB];
            v16bf bh = frag16(rh + lb0, rh + lb1);
            v16bf bl = frag16(rl + lb0, rl + lb1);
            acc[nt] = __builtin_amdgcn_wmma_f32_16x16x32_bf16(false, af, false, bh,
                                                              (short)0, acc[nt], false, false);
            acc[nt] = __builtin_amdgcn_wmma_f32_16x16x32_bf16(false, af, false, bl,
                                                              (short)0, acc[nt], false, false);
        }
        __syncthreads();                                // buffer free before next TDM overwrite
    }
#else
    // Fallback: cooperative global->VGPR->LDS staging (single buffer).
    __shared__ __align__(16) __bf16 sBh[64 * LDB];
    __shared__ __align__(16) __bf16 sBl[64 * LDB];
    const int ldRow = threadIdx.x >> 2;
    const int ldCol = (threadIdx.x & 3) * 8;
    const __bf16* gBh = Bhi + (size_t)(nBase + ldRow) * K + ldCol;
    const __bf16* gBl = Blo + (size_t)(nBase + ldRow) * K + ldCol;
    for (int kc = 0; kc < K; kc += 32) {
        *(v8bf*)&sBh[ldRow * LDB + ldCol] = *(const v8bf*)(gBh + kc);
        *(v8bf*)&sBl[ldRow * LDB + ldCol] = *(const v8bf*)(gBl + kc);
        __syncthreads();
        const int kb0 = kc + hiHalf * 8, kb1 = kc + 16 + hiHalf * 8;
        v16bf af = frag16(A + kb0, A + kb1);
        if (kc + 32 < K) __builtin_prefetch(A + kc + 32, 0, 0);
        const int lb0 = hiHalf * 8, lb1 = 16 + hiHalf * 8;
#pragma unroll
        for (int nt = 0; nt < 4; ++nt) {
            const __bf16* rh = &sBh[(nt * 16 + laneN) * LDB];
            const __bf16* rl = &sBl[(nt * 16 + laneN) * LDB];
            v16bf bh = frag16(rh + lb0, rh + lb1);
            v16bf bl = frag16(rl + lb0, rl + lb1);
            acc[nt] = __builtin_amdgcn_wmma_f32_16x16x32_bf16(false, af, false, bh,
                                                              (short)0, acc[nt], false, false);
            acc[nt] = __builtin_amdgcn_wmma_f32_16x16x32_bf16(false, af, false, bl,
                                                              (short)0, acc[nt], false, false);
        }
        __syncthreads();
    }
#endif
    float* C = Call + (size_t)t * M * N;
#pragma unroll
    for (int nt = 0; nt < 4; ++nt) {
        const int n = nBase + nt * 16 + laneN;
        const float bv = bias[n];
#pragma unroll
        for (int r = 0; r < 8; ++r) {
            const int m = mBase + wave * 16 + hiHalf * 8 + r;
            C[(size_t)m * N + n] = acc[nt][r] + bv;
        }
    }
}

// ---------------------------------------------------------------- LIF scan (writes spikes as bf16)
__global__ void lif_scan_kernel(const float* __restrict__ cur, __bf16* __restrict__ spk,
                                int n, int perStep) {
    int i = blockIdx.x * blockDim.x + threadIdx.x;
    if (i >= n) return;
    float mem = 0.f;
#pragma unroll
    for (int t = 0; t < NT; ++t) {
        float c = perStep ? cur[(size_t)t * n + i] : cur[i];
        float reset = (mem - 1.0f) > 0.f ? 1.0f : 0.0f;
        mem = 0.95f * mem + c - reset;
        spk[(size_t)t * n + i] = (__bf16)(((mem - 1.0f) > 0.f) ? 1.0f : 0.0f);
    }
}

// ---------------------------------------------------------------- head: tiny GEMM (O=2) + output LIF
__global__ __launch_bounds__(256)
void head_kernel(const __bf16* __restrict__ spk2,   // [NT, BB, H2]
                 const float* __restrict__ Wout,    // [2, H2]
                 const float* __restrict__ bout,    // [2]
                 float* __restrict__ outSpk,        // [NT, BB, 2]
                 float* __restrict__ outMem) {      // [NT, BB, 2]
    const int wave = threadIdx.x >> 5, lane = threadIdx.x & 31;
    const int b = blockIdx.x * 8 + wave;
    float m0 = 0.f, m1 = 0.f;
    const float b0v = bout[0], b1v = bout[1];
    for (int t = 0; t < NT; ++t) {
        const __bf16* s = spk2 + ((size_t)t * BB + b) * H2;
        float d0 = 0.f, d1 = 0.f;
        for (int k = lane; k < H2; k += 32) {
            float sv = (float)s[k];
            d0 += sv * Wout[k];
            d1 += sv * Wout[H2 + k];
        }
#pragma unroll
        for (int off = 16; off; off >>= 1) {
            d0 += __shfl_xor(d0, off, 32);
            d1 += __shfl_xor(d1, off, 32);
        }
        d0 += b0v; d1 += b1v;
        float r0 = (m0 - 1.f) > 0.f ? 1.f : 0.f;
        float r1 = (m1 - 1.f) > 0.f ? 1.f : 0.f;
        m0 = 0.95f * m0 + d0 - r0;
        m1 = 0.95f * m1 + d1 - r1;
        float s0 = (m0 - 1.f) > 0.f ? 1.f : 0.f;
        float s1 = (m1 - 1.f) > 0.f ? 1.f : 0.f;
        if (lane == 0) {
            size_t o = ((size_t)t * BB + b) * 2;
            outSpk[o] = s0; outSpk[o + 1] = s1;
            outMem[o] = m0; outMem[o + 1] = m1;
        }
    }
}

// ---------------------------------------------------------------- launch
extern "C" void kernel_launch(void* const* d_in, const int* in_sizes, int n_in,
                              void* d_out, int out_size, void* d_ws, size_t ws_size,
                              hipStream_t stream) {
    const float* x    = (const float*)d_in[0];
    const float* W1   = (const float*)d_in[1];
    const float* b1   = (const float*)d_in[2];
    const float* W2   = (const float*)d_in[3];
    const float* b2   = (const float*)d_in[4];
    const float* Wout = (const float*)d_in[5];
    const float* bout = (const float*)d_in[6];
    float* out = (float*)d_out;

    char* ws = (char*)d_ws;
    size_t off = 0;
    auto alloc = [&](size_t bytes) -> void* {
        void* p = ws + off;
        off += (bytes + 255) & ~(size_t)255;
        return p;
    };
    // ~440 MB of scratch total
    __bf16* xh  = (__bf16*)alloc((size_t)BB * DIN * 2);
    __bf16* xl  = (__bf16*)alloc((size_t)BB * DIN * 2);
    __bf16* w1h = (__bf16*)alloc((size_t)H1 * DIN * 2);
    __bf16* w1l = (__bf16*)alloc((size_t)H1 * DIN * 2);
    __bf16* w2h = (__bf16*)alloc((size_t)H2 * H1 * 2);
    __bf16* w2l = (__bf16*)alloc((size_t)H2 * H1 * 2);
    float*  cur1 = (float*)alloc((size_t)BB * H1 * 4);
    __bf16* spk1 = (__bf16*)alloc((size_t)NT * BB * H1 * 2);
    float*  cur2 = (float*)alloc((size_t)NT * BB * H2 * 4);
    __bf16* spk2 = (__bf16*)alloc((size_t)NT * BB * H2 * 2);

    // 1) hi/lo splits
    {
        int n = BB * DIN;
        split_hilo_kernel<<<(n + 255) / 256, 256, 0, stream>>>(x, xh, xl, n);
        n = H1 * DIN;
        split_hilo_kernel<<<(n + 255) / 256, 256, 0, stream>>>(W1, w1h, w1l, n);
        n = H2 * H1;
        split_hilo_kernel<<<(n + 255) / 256, 256, 0, stream>>>(W2, w2h, w2l, n);
    }
    // 2) cur1 = x @ W1^T + b1   (M=1024, N=4096, K=2048)
    gemm_hilo3_kernel<<<dim3(H1 / 64, BB / 128, 1), 256, 0, stream>>>(
        xh, xl, w1h, w1l, b1, cur1, BB, H1, DIN);
    // 3) LIF layer 1 -> spk1[t]
    {
        int n = BB * H1;
        lif_scan_kernel<<<(n + 255) / 256, 256, 0, stream>>>(cur1, spk1, n, /*perStep=*/0);
    }
    // 4) cur2[t] = spk1[t] @ W2^T + b2   (M=1024, N=1024, K=4096, t=0..24)
    gemm_spk2_kernel<<<dim3(H2 / 64, BB / 128, NT), 256, 0, stream>>>(
        spk1, w2h, w2l, b2, cur2, BB, H2, H1);
    // 5) LIF layer 2 -> spk2[t]
    {
        int n = BB * H2;
        lif_scan_kernel<<<(n + 255) / 256, 256, 0, stream>>>(cur2, spk2, n, /*perStep=*/1);
    }
    // 6) head + output LIF; out = [spk_rec | mem_rec], each NT*BB*2 floats
    head_kernel<<<BB / 8, 256, 0, stream>>>(spk2, Wout, bout,
                                            out, out + (size_t)NT * BB * 2);
}
